// CaMoE_Block_75831942578808
// MI455X (gfx1250) — compile-verified, matrix operand order checked by reading, not run on hardware
//
#include <hip/hip_runtime.h>
#include <hip/hip_bf16.h>
#include <math.h>

// ---------------- problem constants ----------------
#define C_DIM 1024
#define H_DIM 4096
#define NTOK  8192          // B*T = 4*2048
#define NE    8             // total experts
#define NR_E  6             // rwkv experts

typedef __attribute__((ext_vector_type(16))) _Float16 v16h;
typedef __attribute__((ext_vector_type(8)))  _Float16 v8h;
typedef __attribute__((ext_vector_type(8)))  float    v8f;

__device__ __forceinline__ float sigmf(float x) { return 1.0f / (1.0f + __expf(-x)); }
__device__ __forceinline__ int   imin(int a, int b) { return a < b ? a : b; }

// CDNA5 async global->LDS DMA (VGLOBAL op 98), tracked with ASYNCcnt.
// the LDS address is taken from VDST; generic pointers to __shared__ carry the
// LDS byte offset in their low 32 bits, so truncation gives VDST's value.
__device__ __forceinline__ void async_load_b128(void* lds_ptr, const void* gptr) {
    unsigned int lds_off = (unsigned int)(unsigned long long)lds_ptr;
    asm volatile("global_load_async_to_lds_b128 %0, %1, off"
                 :: "v"(lds_off), "v"(gptr) : "memory");
}
__device__ __forceinline__ void wait_async0() {
    asm volatile("s_wait_asynccnt 0x0" ::: "memory");
}

// =====================================================================
// Generic tiled WMMA GEMM:  C[M,N] (+)= act( A16[M,K] @ B16[K,N] )
// Block tile 128x128, K-step 32, 256 threads = 8 waves, wave tile 32x64.
// A tile staged with GLOBAL_LOAD_ASYNC_TO_LDS_B128 (ASYNCcnt), B tile
// transposed through VGPRs so fragment reads are contiguous ds_load_b128.
// EPI: 0 store f32, 1 accumulate f32, 2 sigmoid f32, 3 relu^2 -> f16,
//      4 gelu -> f16.
// GATHER: A row indices come from gidx[] (sparse expert dispatch).
// cntp: device row count (compact M); nullptr -> count = M.
// =====================================================================
#define BM 128
#define BN 128
#define BK 32
#define LDP 40   // LDS row pitch in halves (padded: 80B, 16B aligned)

template <int EPI, bool GATHER>
__global__ __launch_bounds__(256) void wmma_gemm_kernel(
    const _Float16* __restrict__ A, const _Float16* __restrict__ B,
    void* __restrict__ Cp, int M, int N, int K,
    const int* __restrict__ gidx, const int* __restrict__ cntp)
{
    const int tid  = threadIdx.x;
    const int lane = tid & 31;
    const int wave = tid >> 5;
    const int wm   = wave >> 1;            // 0..3  (rows of 32)
    const int wn   = wave & 1;             // 0..1  (cols of 64)
    const int nbase = blockIdx.x * BN;
    const int mbase = blockIdx.y * BM;
    const int count = cntp ? *cntp : M;
    if (mbase >= count) return;

    __shared__ _Float16 As[BM * LDP];      // [128 rows][32 K] padded
    __shared__ _Float16 Bt[BN * LDP];      // transposed: [128 cols][32 K]

    // --- per-thread global load coordinates (fixed across K loop) ---
    const int ar = tid >> 1;               // 0..127 A tile row
    const int ac = (tid & 1) * 16;         // 0 or 16 (halves)
    int arow_g = imin(mbase + ar, count - 1);
    if (GATHER) arow_g = gidx[arow_g];
    const _Float16* aptr = A + (size_t)arow_g * K;
    _Float16* as_dst = &As[ar * LDP + ac];

    const int br = tid >> 3;               // 0..31  B tile K-row
    const int bc = (tid & 7) * 16;         // 0..112 B tile col (halves)
    const _Float16* bptr = B + (size_t)br * N + nbase + bc;

    v8f acc[2][4];
    #pragma unroll
    for (int i = 0; i < 2; ++i)
        #pragma unroll
        for (int j = 0; j < 4; ++j)
            #pragma unroll
            for (int r = 0; r < 8; ++r) acc[i][j][r] = 0.0f;

    for (int ks = 0; ks < K; ks += BK) {
        __syncthreads();   // previous iteration's fragment reads done

        // ---- A tile: 2x async global->LDS b128 DMAs per thread ----
        async_load_b128(as_dst,     aptr + ks + ac);
        async_load_b128(as_dst + 8, aptr + ks + ac + 8);
        if (ks + BK < K) __builtin_prefetch(aptr + ks + BK + ac, 0, 1);

        // ---- B tile transposed: b128 loads -> 16x b16 LDS stores ----
        {
            const float4* gb = (const float4*)(bptr + (size_t)ks * N);
            float4 b0 = gb[0];
            float4 b1 = gb[1];
            _Float16 hb[16];
            *(float4*)(hb)     = b0;
            *(float4*)(hb + 8) = b1;
            #pragma unroll
            for (int i = 0; i < 16; ++i)
                Bt[(bc + i) * LDP + br] = hb[i];
            if (ks + BK < K) __builtin_prefetch(bptr + (size_t)(ks + BK) * N, 0, 1);
        }
        wait_async0();     // this wave's async LDS writes are visible
        __syncthreads();   // all waves' tiles published

        // ---- load fragments from LDS (16-bit A 16x32 lane layout) ----
        const int koff = (lane >> 4) * 8;  // lane half selects K sub-block
        v16h av[2], bv[4];
        #pragma unroll
        for (int fm = 0; fm < 2; ++fm) {
            const int arow = wm * 32 + fm * 16 + (lane & 15);
            v8h lo = *(const v8h*)(&As[arow * LDP + koff]);
            v8h hi = *(const v8h*)(&As[arow * LDP + 16 + koff]);
            av[fm] = __builtin_shufflevector(lo, hi, 0,1,2,3,4,5,6,7,8,9,10,11,12,13,14,15);
        }
        #pragma unroll
        for (int fn = 0; fn < 4; ++fn) {
            const int bcol = wn * 64 + fn * 16 + (lane & 15);
            v8h lo = *(const v8h*)(&Bt[bcol * LDP + koff]);
            v8h hi = *(const v8h*)(&Bt[bcol * LDP + 16 + koff]);
            bv[fn] = __builtin_shufflevector(lo, hi, 0,1,2,3,4,5,6,7,8,9,10,11,12,13,14,15);
        }

        // ---- 8x v_wmma_f32_16x16x32_f16 per wave per K-step ----
        #pragma unroll
        for (int fm = 0; fm < 2; ++fm)
            #pragma unroll
            for (int fn = 0; fn < 4; ++fn)
                acc[fm][fn] = __builtin_amdgcn_wmma_f32_16x16x32_f16(
                    false, av[fm], false, bv[fn], (short)0, acc[fm][fn],
                    false, false);
    }

    // ---- epilogue: C/D layout -> VGPR r holds row (lane>=16 ? 8+r : r) ----
    const int colb = nbase + wn * 64 + (lane & 15);
    const int rowb = mbase + wm * 32 + ((lane >> 4) << 3);
    #pragma unroll
    for (int fm = 0; fm < 2; ++fm)
        #pragma unroll
        for (int fn = 0; fn < 4; ++fn)
            #pragma unroll
            for (int r = 0; r < 8; ++r) {
                const int row = rowb + fm * 16 + r;
                const int col = colb + fn * 16;
                if (row < count && col < N) {
                    const size_t off = (size_t)row * N + col;
                    float v = acc[fm][fn][r];
                    if (EPI == 0)      ((float*)Cp)[off] = v;
                    else if (EPI == 1) ((float*)Cp)[off] += v;
                    else if (EPI == 2) ((float*)Cp)[off] = sigmf(v);
                    else if (EPI == 3) { float t = v > 0.0f ? v : 0.0f;
                                         ((_Float16*)Cp)[off] = (_Float16)(t * t); }
                    else { float g = 0.5f * v * (1.0f + tanhf(0.7978845608f *
                                     (v + 0.044715f * v * v * v)));
                           ((_Float16*)Cp)[off] = (_Float16)g; }
                }
            }
}

// =====================================================================
// Elementwise / reduction kernels
// =====================================================================
__global__ void cvt_f16_kernel(const float* __restrict__ s,
                               _Float16* __restrict__ d, int n) {
    int i = blockIdx.x * blockDim.x + threadIdx.x;
    if (i < n) d[i] = (_Float16)s[i];
}

__global__ void copy_f32_kernel(const float* __restrict__ s,
                                float* __restrict__ d, int n) {
    int i = blockIdx.x * blockDim.x + threadIdx.x;
    if (i < n) d[i] = s[i];
}

__global__ void rv_kernel(const float* __restrict__ r, const float* __restrict__ v,
                          _Float16* __restrict__ o, int n) {
    int i = blockIdx.x * blockDim.x + threadIdx.x;
    if (i < n) o[i] = (_Float16)(sigmf(r[i]) * v[i]);
}

__global__ void add2_f16_kernel(const float* __restrict__ a, const float* __restrict__ b,
                                _Float16* __restrict__ o, int n) {
    int i = blockIdx.x * blockDim.x + threadIdx.x;
    if (i < n) o[i] = (_Float16)(a[i] + b[i]);
}

__global__ void zero_i32_kernel(int* __restrict__ p, int n) {
    int i = blockIdx.x * blockDim.x + threadIdx.x;
    if (i < n) p[i] = 0;
}

// LayerNorm over C=1024; one 256-thread block per token.
__global__ __launch_bounds__(256) void ln_kernel(
    const float* __restrict__ x, const float* __restrict__ add,
    const float* __restrict__ w, const float* __restrict__ b,
    float* __restrict__ sum_out, float* __restrict__ o32,
    _Float16* __restrict__ o16)
{
    const int t = blockIdx.x;
    const int tid = threadIdx.x;
    __shared__ float sh[C_DIM];
    __shared__ float red[256];
    const size_t base = (size_t)t * C_DIM;

    float ls = 0.0f, lq = 0.0f;
    #pragma unroll
    for (int j = 0; j < 4; ++j) {
        int c = tid + j * 256;
        float v = x[base + c];
        if (add) v += add[base + c];
        sh[c] = v;
        if (sum_out) sum_out[base + c] = v;
        ls += v; lq += v * v;
    }
    red[tid] = ls; __syncthreads();
    for (int s = 128; s > 0; s >>= 1) {
        if (tid < s) red[tid] += red[tid + s];
        __syncthreads();
    }
    float mean = red[0] * (1.0f / C_DIM);
    __syncthreads();
    red[tid] = lq; __syncthreads();
    for (int s = 128; s > 0; s >>= 1) {
        if (tid < s) red[tid] += red[tid + s];
        __syncthreads();
    }
    float var = red[0] * (1.0f / C_DIM) - mean * mean;
    float rstd = rsqrtf(var + 1e-5f);
    #pragma unroll
    for (int j = 0; j < 4; ++j) {
        int c = tid + j * 256;
        float v = (sh[c] - mean) * rstd * w[c] + b[c];
        if (o32) o32[base + c] = v;
        o16[base + c] = (_Float16)v;
    }
}

// Router: per token computes 17 dot products (8 confidences, 8 affinities,
// 1 difficulty), top-2 with softmax weights, writes outputs and builds
// per-expert compact token lists with atomics.
__global__ __launch_bounds__(256) void routing_kernel(
    const float* __restrict__ h,
    const float* __restrict__ conf_r,   // [6,C]
    const float* __restrict__ conf_t,   // [2,C]
    const float* __restrict__ Wa,       // [C,8]
    const float* __restrict__ Wd,       // [C,1]
    const float* __restrict__ capital,  // [8]
    float* __restrict__ o_win, float* __restrict__ o_cost,
    float* __restrict__ o_diff, float* __restrict__ o_aff,
    int* __restrict__ ecnt, int* __restrict__ eidx, float* __restrict__ egate)
{
    const int t = blockIdx.x;
    const int tid = threadIdx.x;
    const int lane = tid & 31;
    const int wave = tid >> 5;
    __shared__ float sh[C_DIM];
    __shared__ float dots[24];
    const size_t base = (size_t)t * C_DIM;

    #pragma unroll
    for (int j = 0; j < 4; ++j) { int c = tid + j * 256; sh[c] = h[base + c]; }
    __syncthreads();

    for (int j = 0; j < 3; ++j) {
        int d = wave * 3 + j;
        if (d < 17) {
            float s = 0.0f;
            if (d < 6) {
                const float* w = conf_r + d * C_DIM;
                for (int c = lane; c < C_DIM; c += 32) s += sh[c] * w[c];
            } else if (d < 8) {
                const float* w = conf_t + (d - 6) * C_DIM;
                for (int c = lane; c < C_DIM; c += 32) s += sh[c] * w[c];
            } else if (d < 16) {
                int e = d - 8;
                for (int c = lane; c < C_DIM; c += 32) s += sh[c] * Wa[c * NE + e];
            } else {
                for (int c = lane; c < C_DIM; c += 32) s += sh[c] * Wd[c];
            }
            for (int off = 16; off > 0; off >>= 1) s += __shfl_down(s, off, 32);
            if (lane == 0) dots[d] = s;
        }
    }
    __syncthreads();

    if (tid == 0) {
        float diff = log1pf(__expf(dots[16]));  // softplus
        o_diff[t] = diff;
        float eff[NE];
        #pragma unroll
        for (int e = 0; e < NE; ++e) {
            float conf = sigmf(dots[e]);
            float aff  = dots[8 + e];
            o_aff[(size_t)t * NE + e] = aff;
            eff[e] = (conf * capital[e] + aff) * diff;
        }
        int i0 = 0;
        #pragma unroll
        for (int e = 1; e < NE; ++e) if (eff[e] > eff[i0]) i0 = e;
        int i1 = (i0 == 0) ? 1 : 0;
        #pragma unroll
        for (int e = 0; e < NE; ++e) if (e != i0 && eff[e] > eff[i1]) i1 = e;
        float v0 = eff[i0], v1 = eff[i1];
        float m = fmaxf(v0, v1);
        float e0 = __expf(v0 - m), e1 = __expf(v1 - m);
        float z = e0 + e1;
        float w0 = e0 / z, w1 = e1 / z;
        o_win[t * 2 + 0] = (float)i0;
        o_win[t * 2 + 1] = (float)i1;
        o_cost[t] = v0 * w0 + v1 * w1;
        int p0 = atomicAdd(&ecnt[i0], 1);
        eidx[i0 * NTOK + p0] = t; egate[i0 * NTOK + p0] = w0;
        int p1 = atomicAdd(&ecnt[i1], 1);
        eidx[i1 * NTOK + p1] = t; egate[i1 * NTOK + p1] = w1;
    }
}

// Gated scatter-combine:  xacc[tok] += gate * (sig ? sig*kv : kv)
__global__ __launch_bounds__(256) void combine_kernel(
    float* __restrict__ xacc, const float* __restrict__ sig,
    const float* __restrict__ kv, const int* __restrict__ eidx,
    const float* __restrict__ egate, const int* __restrict__ cntp)
{
    int gid = blockIdx.x * 256 + threadIdx.x;
    int r = gid >> 10;
    int c = gid & (C_DIM - 1);
    if (r >= *cntp) return;
    int t = eidx[r];
    float v = kv[(size_t)r * C_DIM + c];
    if (sig) v *= sig[(size_t)r * C_DIM + c];
    xacc[(size_t)t * C_DIM + c] += egate[r] * v;
}

// =====================================================================
// Host-side orchestration
// =====================================================================
extern "C" void kernel_launch(void* const* d_in, const int* in_sizes, int n_in,
                              void* d_out, int out_size, void* d_ws, size_t ws_size,
                              hipStream_t stream)
{
    const float* x       = (const float*)d_in[0];
    const float* v_first = (const float*)d_in[1];
    const float* capital = (const float*)d_in[2];
    const float* ln1w    = (const float*)d_in[3];
    const float* ln1b    = (const float*)d_in[4];
    const float* ln2w    = (const float*)d_in[5];
    const float* ln2b    = (const float*)d_in[6];
    const float* W_attr  = (const float*)d_in[7];
    const float* W_attv  = (const float*)d_in[8];
    const float* W_atto  = (const float*)d_in[9];
    const float* W_atts  = (const float*)d_in[10];
    const float* Wd      = (const float*)d_in[11];
    const float* Wa      = (const float*)d_in[12];
    const float* Wb_h    = (const float*)d_in[13];
    const float* Wb_s    = (const float*)d_in[14];
    const float* Wk      = (const float*)d_in[15];
    const float* Wv      = (const float*)d_in[16];
    const float* Wr      = (const float*)d_in[17];
    const float* conf_r  = (const float*)d_in[18];
    const float* Wt_in   = (const float*)d_in[19];
    const float* Wt_out  = (const float*)d_in[20];
    const float* conf_t  = (const float*)d_in[21];

    // output layout (flat concat, all as float)
    float* out    = (float*)d_out;
    float* o_x    = out;                          // [NTOK, C]
    float* o_vf   = o_x   + (size_t)NTOK * C_DIM;
    float* o_win  = o_vf  + (size_t)NTOK * C_DIM; // [NTOK, 2]
    float* o_cost = o_win + (size_t)NTOK * 2;     // [NTOK]
    float* o_diff = o_cost + NTOK;                // [NTOK]
    float* o_aff  = o_diff + NTOK;                // [NTOK, 8]

    // workspace bump allocator
    char* ws = (char*)d_ws;
    size_t cur = 0;
    auto alloc = [&](size_t bytes) -> void* {
        void* p = ws + cur;
        cur += (bytes + 255) & ~(size_t)255;
        return p;
    };
    const size_t CC = (size_t)C_DIM * C_DIM;     // 1M
    const size_t CH = (size_t)C_DIM * H_DIM;     // 4M
    const size_t TC = (size_t)NTOK * C_DIM;      // 8M
    const size_t TH = (size_t)NTOK * H_DIM;      // 32M

    // f16 weight copies
    _Float16* wAttr16 = (_Float16*)alloc(CC * 2);
    _Float16* wAttv16 = (_Float16*)alloc(CC * 2);
    _Float16* wAtto16 = (_Float16*)alloc(CC * 2);
    _Float16* wAtts16 = (_Float16*)alloc(CC * 2);
    _Float16* wBh16   = (_Float16*)alloc(CC * 2);
    _Float16* wBs16   = (_Float16*)alloc(CC * 2);
    _Float16* wK16    = (_Float16*)alloc(NR_E * CH * 2);
    _Float16* wV16    = (_Float16*)alloc(NR_E * CH * 2);
    _Float16* wR16    = (_Float16*)alloc(NR_E * CC * 2);
    _Float16* wTi16   = (_Float16*)alloc(2 * CH * 2);
    _Float16* wTo16   = (_Float16*)alloc(2 * CH * 2);
    // activations
    _Float16* xln16   = (_Float16*)alloc(TC * 2);
    _Float16* rv16    = (_Float16*)alloc(TC * 2);
    _Float16* s16     = (_Float16*)alloc(TC * 2);
    _Float16* h16     = (_Float16*)alloc(TC * 2);
    _Float16* hb16    = (_Float16*)alloc(TC * 2);
    _Float16* k2_16   = (_Float16*)alloc(TH * 2);
    float* r_lin   = (float*)alloc(TC * 4);
    float* v_lin   = (float*)alloc(TC * 4);
    float* att_out = (float*)alloc(TC * 4);
    float* s32     = (float*)alloc(TC * 4);
    float* h32     = (float*)alloc(TC * 4);
    float* bridge  = (float*)alloc(TC * 4);
    float* kv32    = (float*)alloc(TC * 4);
    float* sig32   = (float*)alloc(TC * 4);
    // routing state
    int*   ecnt  = (int*)alloc(NE * 4);
    int*   eidx  = (int*)alloc((size_t)NE * NTOK * 4);
    float* egate = (float*)alloc((size_t)NE * NTOK * 4);

    auto cvt = [&](const float* s, _Float16* d, size_t n) {
        cvt_f16_kernel<<<dim3((unsigned)((n + 255) / 256)), dim3(256), 0, stream>>>(s, d, (int)n);
    };

    // ---- stage all weights to f16 ----
    cvt(W_attr, wAttr16, CC);  cvt(W_attv, wAttv16, CC);
    cvt(W_atto, wAtto16, CC);  cvt(W_atts, wAtts16, CC);
    cvt(Wb_h,   wBh16,   CC);  cvt(Wb_s,   wBs16,   CC);
    cvt(Wk, wK16, NR_E * CH);  cvt(Wv, wV16, NR_E * CH);
    cvt(Wr, wR16, NR_E * CC);
    cvt(Wt_in, wTi16, 2 * CH); cvt(Wt_out, wTo16, 2 * CH);

    const dim3 blk(256);
    const dim3 gN1(C_DIM / BN, NTOK / BM);   // N=1024
    const dim3 gN4(H_DIM / BN, NTOK / BM);   // N=4096
    const unsigned gTC = (unsigned)((TC + 255) / 256);

    // ---- attention ----
    ln_kernel<<<NTOK, blk, 0, stream>>>(x, nullptr, ln1w, ln1b,
                                        nullptr, nullptr, xln16);
    wmma_gemm_kernel<0, false><<<gN1, blk, 0, stream>>>(
        xln16, wAttr16, r_lin, NTOK, C_DIM, C_DIM, nullptr, nullptr);
    wmma_gemm_kernel<0, false><<<gN1, blk, 0, stream>>>(
        xln16, wAttv16, v_lin, NTOK, C_DIM, C_DIM, nullptr, nullptr);
    rv_kernel<<<gTC, blk, 0, stream>>>(r_lin, v_lin, rv16, (int)TC);
    wmma_gemm_kernel<0, false><<<gN1, blk, 0, stream>>>(
        rv16, wAtto16, att_out, NTOK, C_DIM, C_DIM, nullptr, nullptr);
    wmma_gemm_kernel<0, false><<<gN1, blk, 0, stream>>>(
        xln16, wAtts16, s32, NTOK, C_DIM, C_DIM, nullptr, nullptr);
    cvt_f16_kernel<<<gTC, blk, 0, stream>>>(s32, s16, (int)TC);

    // x1 = x + att_out (stored in o_x), h = LN2(x1)
    ln_kernel<<<NTOK, blk, 0, stream>>>(x, att_out, ln2w, ln2b,
                                        o_x, h32, h16);

    // ---- bridge = h@Wb_h + s@Wb_s ;  hb = f16(h + bridge) ----
    wmma_gemm_kernel<0, false><<<gN1, blk, 0, stream>>>(
        h16, wBh16, bridge, NTOK, C_DIM, C_DIM, nullptr, nullptr);
    wmma_gemm_kernel<1, false><<<gN1, blk, 0, stream>>>(
        s16, wBs16, bridge, NTOK, C_DIM, C_DIM, nullptr, nullptr);
    add2_f16_kernel<<<gTC, blk, 0, stream>>>(h32, bridge, hb16, (int)TC);

    // ---- routing ----
    zero_i32_kernel<<<1, 32, 0, stream>>>(ecnt, NE);
    routing_kernel<<<NTOK, blk, 0, stream>>>(h32, conf_r, conf_t, Wa, Wd,
        capital, o_win, o_cost, o_diff, o_aff, ecnt, eidx, egate);

    const unsigned gComb = (unsigned)(TC / 256);

    // ---- RWKV experts: out = sigmoid(h@Wr) * (relu(h@Wk)^2 @ Wv) ----
    for (int e = 0; e < NR_E; ++e) {
        const int* idx = eidx + (size_t)e * NTOK;
        const float* gate = egate + (size_t)e * NTOK;
        const int* cnt = ecnt + e;
        wmma_gemm_kernel<3, true><<<gN4, blk, 0, stream>>>(
            h16, wK16 + (size_t)e * CH, k2_16, NTOK, H_DIM, C_DIM, idx, cnt);
        wmma_gemm_kernel<0, false><<<gN1, blk, 0, stream>>>(
            k2_16, wV16 + (size_t)e * CH, kv32, NTOK, C_DIM, H_DIM, nullptr, cnt);
        wmma_gemm_kernel<2, true><<<gN1, blk, 0, stream>>>(
            h16, wR16 + (size_t)e * CC, sig32, NTOK, C_DIM, C_DIM, idx, cnt);
        combine_kernel<<<gComb, blk, 0, stream>>>(o_x, sig32, kv32, idx, gate, cnt);
    }

    // ---- transformer experts: out = gelu((h+bridge)@Wt_in) @ Wt_out ----
    for (int i = 0; i < 2; ++i) {
        const int e = NR_E + i;
        const int* idx = eidx + (size_t)e * NTOK;
        const float* gate = egate + (size_t)e * NTOK;
        const int* cnt = ecnt + e;
        wmma_gemm_kernel<4, true><<<gN4, blk, 0, stream>>>(
            hb16, wTi16 + (size_t)i * CH, k2_16, NTOK, H_DIM, C_DIM, idx, cnt);
        wmma_gemm_kernel<0, false><<<gN1, blk, 0, stream>>>(
            k2_16, wTo16 + (size_t)i * CH, kv32, NTOK, C_DIM, H_DIM, nullptr, cnt);
        combine_kernel<<<gComb, blk, 0, stream>>>(o_x, nullptr, kv32, idx, gate, cnt);
    }

    // ---- v_first passthrough ----
    copy_f32_kernel<<<gTC, blk, 0, stream>>>(v_first, o_vf, (int)TC);
}
